// CrossWindowAttention3D_44607530336778
// MI455X (gfx1250) — matrix operand order repaired and checked by reading, not compiled, not run on hardware
//
#include <hip/hip_runtime.h>
#include <hip/hip_bf16.h>
#include <stdint.h>

// ---------------------------------------------------------------------------
// 3D shifted-window attention (Video-Swin style) on gfx1250 via f16 WMMA.
//   S=48^3 voxels, C=96, 4 heads x hd=24, window 6^3 -> n=216 tokens, 512 wins
// All GEMMs run on v_wmma_f32_16x16x32_f16. Intermediates kept in f16.
// Attention K/V tiles staged to LDS via GLOBAL_LOAD_ASYNC_TO_LDS_B128 when the
// toolchain exposes the gfx1250 async builtins (fallback: plain copies).
// ---------------------------------------------------------------------------

typedef _Float16 h16 __attribute__((ext_vector_type(16)));
typedef _Float16 h2  __attribute__((ext_vector_type(2)));
typedef float    f8  __attribute__((ext_vector_type(8)));

#define VOX   110592         // 48^3
#define C_    96
#define NHEAD 4
#define HD    24
#define NTOK  216            // 6^3 tokens per window
#define NPAD  224            // 14 tiles of 16
#define HDP   32             // hd padded to one K=32 WMMA step
#define MAT   (NPAD*HDP)     // 7168 f16 elements per Q/K/Vt matrix
#define WINS  512            // 8^3 windows
#define NEG_BIG (-3.0e38f)

#if defined(__gfx1250__) && \
    __has_builtin(__builtin_amdgcn_global_load_async_to_lds_b128) && \
    __has_builtin(__builtin_amdgcn_s_wait_asynccnt)
#define USE_ASYNC_LDS 1
// exact pointee type per hipcc diagnostic: int __attribute__((vector_size(16)))
typedef int v4i __attribute__((vector_size(16)));
typedef __attribute__((address_space(1))) v4i gv4i;   // global
typedef __attribute__((address_space(3))) v4i lv4i;   // LDS
#else
#define USE_ASYNC_LDS 0
#endif

__device__ __forceinline__ f8 wmma16(h16 a, h16 b, f8 c) {
  return __builtin_amdgcn_wmma_f32_16x16x32_f16(
      /*neg_a=*/false, a, /*neg_b=*/false, b,
      /*c_mod=*/(short)0, c, /*reuse_a=*/false, /*reuse_b=*/false);
}

// A-matrix fragment (16x32 f16, contiguous K order matching B):
//   lane<16 : row=lane,  K = kb..kb+15, kb=0
//   lane>=16: row=lane-16, same with kb=16
__device__ __forceinline__ h16 load_afrag(const _Float16* base, int rowStride,
                                          int row, int kbase, int lane) {
  h16 a;
  const int kb = (lane >> 4) * 16;
  const _Float16* p = base + (size_t)row * rowStride + kbase + kb;
#pragma unroll
  for (int q = 0; q < 8; ++q) {
    h2 t0 = *(const h2*)(p + 2 * q);          // K = kb+2q, kb+2q+1
    a[2 * q]     = t0[0];
    a[2 * q + 1] = t0[1];
  }
  return a;
}

// B-matrix fragment (32x16 f16): lane col N = lane&15,
//   VGPR v holds K = (lane>=16?16:0) + 2v, 2v+1.
// Source is "N-major" with K contiguous: src[n*rowStride + k].
__device__ __forceinline__ h16 load_bfrag(const _Float16* base, int rowStride,
                                          int ncol, int kbase, int lane) {
  h16 b;
  const int kb = (lane >> 4) * 16;
  const _Float16* p = base + (size_t)ncol * rowStride + kbase + kb;
#pragma unroll
  for (int q = 0; q < 8; ++q) {
    h2 t = *(const h2*)(p + 2 * q);
    b[2 * q]     = t[0];
    b[2 * q + 1] = t[1];
  }
  return b;
}

// ---------------------------------------------------------------------------
__global__ void zero_kernel(uint32_t* __restrict__ p, size_t ndw) {
  size_t i = (size_t)blockIdx.x * blockDim.x + threadIdx.x;
  const size_t stride = (size_t)gridDim.x * blockDim.x;
  for (; i < ndw; i += stride) p[i] = 0u;
}

// ---------------------------------------------------------------------------
// Kernel A: fused QKV 1x1x1-conv (channel GEMM) + shift-roll + window scatter.
// Block = 256 threads (8 waves) = 128 voxels; wave owns a 16-voxel M tile.
__global__ void __launch_bounds__(256)
qkv_proj_kernel(const float* __restrict__ qin, const float* __restrict__ kin,
                const float* __restrict__ vin,
                const float* __restrict__ Wq, const float* __restrict__ bq,
                const float* __restrict__ Wk, const float* __restrict__ bk,
                const float* __restrict__ Wv, const float* __restrict__ bv,
                _Float16* __restrict__ qkv, const int* __restrict__ shiftp) {
  __shared__ _Float16 Wl[C_ * C_];      // 18 KB
  __shared__ _Float16 Xl[128 * C_];     // 24 KB
  const int tid = threadIdx.x;
  const int lane = tid & 31, wave = tid >> 5;
  const int vox0 = blockIdx.x * 128;
  const int s = (*shiftp) ? 3 : 0;
  const int nloc = lane & 15, halfid = lane >> 4;

  // Per-row window id / token id (8 C-fragment rows), independent of q/k/v.
  int winA[8], tokA[8];
#pragma unroll
  for (int g = 0; g < 8; ++g) {
    int vox = vox0 + wave * 16 + g + 8 * halfid;
    int d = vox / 2304; int rem = vox - d * 2304;
    int h = rem / 48;   int w = rem - h * 48;
    int ds = d - s; ds += (ds < 0) ? 48 : 0;     // roll(-s)
    int hs = h - s; hs += (hs < 0) ? 48 : 0;
    int wsX = w - s; wsX += (wsX < 0) ? 48 : 0;
    int wd = ds / 6, td = ds - wd * 6;
    int wh = hs / 6, th = hs - wh * 6;
    int ww = wsX / 6, tw = wsX - ww * 6;
    winA[g] = (wd * 8 + wh) * 8 + ww;
    tokA[g] = (td * 6 + th) * 6 + tw;
  }

  for (int which = 0; which < 3; ++which) {
    const float* W  = (which == 0) ? Wq : ((which == 1) ? Wk : Wv);
    const float* bb = (which == 0) ? bq : ((which == 1) ? bk : bv);
    const float* X  = (which == 0) ? qin : ((which == 1) ? kin : vin);
    __syncthreads();
    for (int idx = tid; idx < C_ * C_; idx += 256)
      Wl[idx] = (_Float16)W[idx];                          // Wl[n*96+k]
    for (int idx = tid; idx < 128 * C_; idx += 256) {
      int ch = idx >> 7, r = idx & 127;                    // coalesced over vox
      Xl[r * C_ + ch] = (_Float16)X[(size_t)ch * VOX + vox0 + r];
    }
    __syncthreads();

    h16 afr[3];
#pragma unroll
    for (int kt = 0; kt < 3; ++kt)
      afr[kt] = load_afrag(Xl, C_, wave * 16 + nloc, kt * 32, lane);

#pragma unroll
    for (int nt = 0; nt < 6; ++nt) {
      const int n = nt * 16 + nloc;
      f8 acc = {};
#pragma unroll
      for (int kt = 0; kt < 3; ++kt) {
        h16 bfr = load_bfrag(Wl, C_, n, kt * 32, lane);    // B[k][n]=W[n][k]
        acc = wmma16(afr[kt], bfr, acc);
      }
      const float biasn = bb[n];
      const int head = n / HD, hc = n - head * HD;
#pragma unroll
      for (int g = 0; g < 8; ++g) {
        float val = acc[g] + biasn;
        if (which == 0) val *= 0.20412414523193154f;       // hd^-0.5 folded in
        const size_t base =
            ((size_t)(winA[g] * NHEAD + head) * 3 + which) * MAT;
        const size_t addr = (which == 2)
            ? (base + (size_t)hc * NPAD + tokA[g])          // V transposed
            : (base + (size_t)tokA[g] * HDP + hc);          // Q/K row-major
        qkv[addr] = (_Float16)val;
      }
    }
  }
}

// ---------------------------------------------------------------------------
// Kernel B: windowed attention. Block = one (window, head), 4 waves; each wave
// owns query row-tiles i = wave, wave+4, ... (14 tiles of 16, N padded 224).
// K [224x32] and V^T [32x224] are staged once per block into LDS (async path).
__global__ void __launch_bounds__(128)
attn_kernel(const _Float16* __restrict__ qkv, const float* __restrict__ rel,
            _Float16* __restrict__ attnb, const int* __restrict__ shiftp) {
  __shared__ _Float16 Kl[MAT];                             // 14 KB [224][32]
  __shared__ _Float16 Vl[MAT];                             // 14 KB [32][224]
  __shared__ _Float16 Pl[4 * 16 * NPAD];                   // 28 KB, wave-private
  const int tid = threadIdx.x, lane = tid & 31, wave = tid >> 5;
  const int win = blockIdx.x >> 2, head = blockIdx.x & 3;
  const int s = (*shiftp) ? 3 : 0;
  const int wd = win >> 6, wh = (win >> 3) & 7, ww = win & 7;
  const int nloc = lane & 15, halfid = lane >> 4;
  const _Float16* Qg = qkv + ((size_t)(win * NHEAD + head) * 3) * MAT;
  const _Float16* Kg = Qg + MAT;
  const _Float16* Vt = Qg + 2 * MAT;                       // [32][224]
  _Float16* Pw = &Pl[wave * 16 * NPAD];

  // ---- stage K and V^T into LDS (2*MAT halfs = 1792 x 16B chunks) ----
#if USE_ASYNC_LDS
  for (int idx = tid; idx < (2 * MAT) / 8; idx += 128) {
    const bool isK = idx < MAT / 8;
    const _Float16* src = isK ? (Kg + idx * 8) : (Vt + (idx - MAT / 8) * 8);
    _Float16* dst       = isK ? (Kl + idx * 8) : (Vl + (idx - MAT / 8) * 8);
    __builtin_amdgcn_global_load_async_to_lds_b128(
        (gv4i*)src, (lv4i*)dst, 0, 0);
  }
  __builtin_amdgcn_s_wait_asynccnt(0);
  __syncthreads();
#else
  for (int idx = tid; idx < (2 * MAT) / 8; idx += 128) {
    const bool isK = idx < MAT / 8;
    const uint4* src = isK ? ((const uint4*)Kg + idx)
                           : ((const uint4*)Vt + (idx - MAT / 8));
    uint4* dst = isK ? ((uint4*)Kl + idx) : ((uint4*)Vl + (idx - MAT / 8));
    *dst = *src;
  }
  __syncthreads();
#endif

  // Key-token coords: fixed per lane per column tile j.
  int tjd[14], tjh[14], tjw[14]; bool jvalid[14];
#pragma unroll
  for (int j = 0; j < 14; ++j) {
    const int tj = j * 16 + nloc;
    jvalid[j] = (tj < NTOK);
    const int dj = tj / 36, rj = tj - dj * 36;
    tjd[j] = dj; tjh[j] = rj / 6; tjw[j] = rj - (rj / 6) * 6;
  }

  for (int i = wave; i < 14; i += 4) {
    // S = (Q*scale) K^T : single K=32 WMMA step (hd zero-padded to 32)
    h16 aq = load_afrag(Qg, HDP, i * 16 + nloc, 0, lane);
    f8 c[14];
#pragma unroll
    for (int j = 0; j < 14; ++j) {
      h16 bkf = load_bfrag(Kl, HDP, j * 16 + nloc, 0, lane);
      f8 z = {};
      c[j] = wmma16(aq, bkf, z);
    }

    // relative-position bias + analytic shift mask
#pragma unroll
    for (int g = 0; g < 8; ++g) {
      const int ti = i * 16 + g + 8 * halfid;
      const int di = ti / 36, ri = ti - di * 36;
      const int hi2 = ri / 6, wi2 = ri - hi2 * 6;
#pragma unroll
      for (int j = 0; j < 14; ++j) {
        if (!jvalid[j]) { c[j][g] = NEG_BIG; continue; }
        const int idx = ((di - tjd[j] + 5) * 11 + (hi2 - tjh[j] + 5)) * 11 +
                        (wi2 - tjw[j] + 5);
        float b = rel[idx * NHEAD + head];
        if (s) {
          const bool diff =
              ((wd == 7) && ((di  < 3) != (tjd[j] < 3))) ||
              ((wh == 7) && ((hi2 < 3) != (tjh[j] < 3))) ||
              ((ww == 7) && ((wi2 < 3) != (tjw[j] < 3)));
          if (diff) b += -100.0f;
        }
        c[j][g] += b;
      }
    }

    // row softmax: row M sits in VGPR (M&7) across one 16-lane half
#pragma unroll
    for (int g = 0; g < 8; ++g) {
      float mx = NEG_BIG;
#pragma unroll
      for (int j = 0; j < 14; ++j) mx = fmaxf(mx, c[j][g]);
      for (int off = 8; off >= 1; off >>= 1)
        mx = fmaxf(mx, __shfl_xor(mx, off, 16));
      float sum = 0.f;
#pragma unroll
      for (int j = 0; j < 14; ++j) {
        float e = __expf(c[j][g] - mx);
        c[j][g] = e; sum += e;
      }
      for (int off = 8; off >= 1; off >>= 1)
        sum += __shfl_xor(sum, off, 16);
      const float rs = 1.0f / sum;
#pragma unroll
      for (int j = 0; j < 14; ++j) c[j][g] *= rs;
    }

    // re-layout P (C-frag -> A-frag) through wave-private LDS
#pragma unroll
    for (int j = 0; j < 14; ++j)
#pragma unroll
      for (int g = 0; g < 8; ++g)
        Pw[(g + 8 * halfid) * NPAD + j * 16 + nloc] = (_Float16)c[j][g];

    // O = P @ V : 7 K-steps of 32, 2 N-tiles (hd padded to 32)
    f8 o0 = {}, o1 = {};
#pragma unroll
    for (int kt = 0; kt < 7; ++kt) {
      h16 ap = load_afrag(Pw, NPAD, nloc, kt * 32, lane);
      h16 b0 = load_bfrag(Vl, NPAD, nloc,      kt * 32, lane);
      h16 b1 = load_bfrag(Vl, NPAD, 16 + nloc, kt * 32, lane);
      o0 = wmma16(ap, b0, o0);
      o1 = wmma16(ap, b1, o1);
    }

    // window-reverse + reverse roll -> spatial f16 buffer [vox][96]
#pragma unroll
    for (int nt = 0; nt < 2; ++nt) {
      const int n = nt * 16 + nloc;
      if (n >= HD) continue;
      f8 o = nt ? o1 : o0;
#pragma unroll
      for (int g = 0; g < 8; ++g) {
        const int t = i * 16 + g + 8 * halfid;
        if (t >= NTOK) continue;
        const int td = t / 36, rt = t - td * 36;
        const int th = rt / 6, tw = rt - th * 6;
        int xd = wd * 6 + td + s; xd -= (xd >= 48) ? 48 : 0;
        int xh = wh * 6 + th + s; xh -= (xh >= 48) ? 48 : 0;
        int xw = ww * 6 + tw + s; xw -= (xw >= 48) ? 48 : 0;
        const int vox = (xd * 48 + xh) * 48 + xw;
        attnb[(size_t)vox * C_ + head * HD + n] = (_Float16)o[g];
      }
    }
  }
}

// ---------------------------------------------------------------------------
// Kernel C: output projection GEMM, f16 A rows contiguous -> half2 frag loads.
__global__ void __launch_bounds__(256)
out_proj_kernel(const _Float16* __restrict__ attnb, const float* __restrict__ Wp,
                const float* __restrict__ bp, float* __restrict__ out) {
  __shared__ _Float16 Wl[C_ * C_];
  const int tid = threadIdx.x, lane = tid & 31, wave = tid >> 5;
  const int vox0 = blockIdx.x * 128;
  const int nloc = lane & 15, halfid = lane >> 4;
  for (int idx = tid; idx < C_ * C_; idx += 256)
    Wl[idx] = (_Float16)Wp[idx];
  __syncthreads();

  const int rowv = vox0 + wave * 16 + nloc;
  h16 afr[3];
#pragma unroll
  for (int kt = 0; kt < 3; ++kt)
    afr[kt] = load_afrag(attnb, C_, rowv, kt * 32, lane);

#pragma unroll
  for (int nt = 0; nt < 6; ++nt) {
    const int n = nt * 16 + nloc;
    f8 acc = {};
#pragma unroll
    for (int kt = 0; kt < 3; ++kt) {
      h16 bfr = load_bfrag(Wl, C_, n, kt * 32, lane);
      acc = wmma16(afr[kt], bfr, acc);
    }
    const float biasn = bp[n];
#pragma unroll
    for (int g = 0; g < 8; ++g) {
      const int vox = vox0 + wave * 16 + g + 8 * halfid;
      out[(size_t)n * VOX + vox] = acc[g] + biasn;   // channel-major output
    }
  }
}

// ---------------------------------------------------------------------------
extern "C" void kernel_launch(void* const* d_in, const int* in_sizes, int n_in,
                              void* d_out, int out_size, void* d_ws,
                              size_t ws_size, hipStream_t stream) {
  (void)in_sizes; (void)n_in; (void)out_size; (void)ws_size;
  const float* qin = (const float*)d_in[0];
  const float* kin = (const float*)d_in[1];
  const float* vin = (const float*)d_in[2];
  const float* Wq  = (const float*)d_in[3];
  const float* bq  = (const float*)d_in[4];
  const float* Wk  = (const float*)d_in[5];
  const float* bk  = (const float*)d_in[6];
  const float* Wv  = (const float*)d_in[7];
  const float* bv  = (const float*)d_in[8];
  const float* Wp  = (const float*)d_in[9];
  const float* bp  = (const float*)d_in[10];
  const float* rel = (const float*)d_in[11];
  const int*   shf = (const int*)d_in[12];

  // Workspace layout: [qkv windowed f16 | spatial attention-out f16]
  _Float16* qkv = (_Float16*)d_ws;
  const size_t QKV_ELEMS = (size_t)WINS * NHEAD * 3 * MAT; // 44,040,192 halfs
  _Float16* attnb = qkv + QKV_ELEMS;

  // zero padded Q/K/Vt tiles (rows 216..223, hd cols 24..31 must be 0)
  zero_kernel<<<2048, 256, 0, stream>>>((uint32_t*)qkv, QKV_ELEMS / 2);

  qkv_proj_kernel<<<VOX / 128, 256, 0, stream>>>(
      qin, kin, vin, Wq, bq, Wk, bk, Wv, bv, qkv, shf);

  attn_kernel<<<WINS * NHEAD, 128, 0, stream>>>(qkv, rel, attnb, shf);

  out_proj_kernel<<<VOX / 128, 256, 0, stream>>>(attnb, Wp, bp, (float*)d_out);
}
